// AudioEmotionModel_26869315404138
// MI455X (gfx1250) — matrix operand order repaired and checked by reading, not compiled, not run on hardware
//
#include <hip/hip_runtime.h>
#include <hip/hip_bf16.h>
#include <math.h>

// ---------------- model constants ----------------
#define BATCH   8
#define TSTEPS  1024
#define PATCH   512
#define HOP     160
#define NSAMP   (PATCH + HOP * (TSTEPS - 1))   // 164192
#define NFREQ   257
#define MROWS   (BATCH * TSTEPS)               // 8192
#define NB_PAD  320                            // DFT basis column pad (1-tile waves)

typedef __attribute__((ext_vector_type(16))) __bf16        v16bf;
typedef __attribute__((ext_vector_type(8)))  float         v8f;
typedef __attribute__((ext_vector_type(8)))  unsigned int  v8u;
typedef __attribute__((ext_vector_type(2)))  float         v2f;
typedef __attribute__((ext_vector_type(2)))  __bf16        v2bf;

#define PI_F 3.14159265358979323846f

static __device__ inline unsigned int pack2bf(float a, float b) {
    v2f x = {a, b};
    v2bf p = __builtin_convertvector(x, v2bf);   // v_cvt_pk_bf16_f32
    return __builtin_bit_cast(unsigned int, p);
}

// ============================================================
// CDNA5 wave32 WMMA fragment layouts (16-bit operands):
//  A (16x32 MxK): lane&15 = M row; VGPR v = K pair
//                 K = (v&4?16:0) + (lane>>4)*8 + (v&3)*2 + p
//  B (32x16 KxN): lane&15 = N, element s -> K as for A  (padded, transposed K-major
//                 weights -> 16 bf16 loads per lane at stride ld)
//  C/D (f32):     lane&15 = N, vgpr r -> M = (lane>>4)*8 + r
// ============================================================

static __device__ inline v16bf load_a_frag(const unsigned int (*AsU)[17], int lane) {
    const int mrow = lane & 15;
    const int half = lane >> 4;
    v8u au;
    #pragma unroll
    for (int v = 0; v < 8; ++v) {
        int cp = ((v & 4) ? 8 : 0) + half * 4 + (v & 3);
        au[v] = AsU[mrow][cp];
    }
    return __builtin_bit_cast(v16bf, au);
}

static __device__ inline v16bf load_b_frag(const __bf16* Bk, int ld, int lane) {
    const int n = lane & 15;
    const int half = lane >> 4;
    v16bf b;
    #pragma unroll
    for (int s = 0; s < 16; ++s) {
        int k = (s < 8) ? (half * 8 + s) : (16 + half * 8 + (s - 8));
        b[s] = Bk[(size_t)k * ld + n];
    }
    return b;
}

// ---------------- generic GEMM: C[M,N] = epi(A[M,K(lda)] * Wt^T) -------------
// Wt bf16 K-major, zero-padded: Kpad mult 32, Npad mult 128.
// Each wave computes two 16x16 tiles (32 N) sharing one A fragment.
// M must be a multiple of 16.  epi: 0 = none, 1 = softplus(x + bias[n])
__global__ __launch_bounds__(128)
void gemm_bf16_wmma(const float* __restrict__ A, const __bf16* __restrict__ Wt,
                    const float* __restrict__ bias, float* __restrict__ C,
                    int M, int N, int K, int lda, int Npad, int epi)
{
    __shared__ unsigned int AsU[16][17];
    const int wave = threadIdx.x >> 5;
    const int lane = threadIdx.x & 31;
    const int m0 = blockIdx.x * 16;
    const int n0 = blockIdx.y * 128 + wave * 32;

    v8f acc0 = {}, acc1 = {};
    const int kfull = K >> 5;

    for (int ks = 0; ks < kfull; ++ks) {               // guard-free body
        const int kb = ks << 5;
        __syncthreads();
        {
            int i = threadIdx.x;
            #pragma unroll
            for (int it = 0; it < 2; ++it, i += 128) {
                int r = i >> 4, cp = i & 15;
                const float* ar = A + (size_t)(m0 + r) * lda + kb + cp * 2;
                AsU[r][cp] = pack2bf(ar[0], ar[1]);
            }
        }
        __syncthreads();
        v16bf af = load_a_frag(AsU, lane);
        const __bf16* wrow = Wt + (size_t)kb * Npad + n0;
        v16bf b0 = load_b_frag(wrow, Npad, lane);
        v16bf b1 = load_b_frag(wrow + 16, Npad, lane);
        acc0 = __builtin_amdgcn_wmma_f32_16x16x32_bf16(false, af, false, b0,
                                                       (short)0, acc0, false, false);
        acc1 = __builtin_amdgcn_wmma_f32_16x16x32_bf16(false, af, false, b1,
                                                       (short)0, acc1, false, false);
    }

    if (K & 31) {                                      // single guarded tail step
        const int kb = kfull << 5;
        __syncthreads();
        {
            int i = threadIdx.x;
            #pragma unroll
            for (int it = 0; it < 2; ++it, i += 128) {
                int r = i >> 4, cp = i & 15;
                int k = kb + cp * 2;
                const float* ar = A + (size_t)(m0 + r) * lda;
                int k0 = k     < K ? k     : K - 1;
                int k1 = k + 1 < K ? k + 1 : K - 1;
                float a0 = ar[k0];
                float a1 = ar[k1];
                a0 = (k     < K) ? a0 : 0.f;
                a1 = (k + 1 < K) ? a1 : 0.f;
                AsU[r][cp] = pack2bf(a0, a1);
            }
        }
        __syncthreads();
        v16bf af = load_a_frag(AsU, lane);
        const __bf16* wrow = Wt + (size_t)kb * Npad + n0;
        v16bf b0 = load_b_frag(wrow, Npad, lane);
        v16bf b1 = load_b_frag(wrow + 16, Npad, lane);
        acc0 = __builtin_amdgcn_wmma_f32_16x16x32_bf16(false, af, false, b0,
                                                       (short)0, acc0, false, false);
        acc1 = __builtin_amdgcn_wmma_f32_16x16x32_bf16(false, af, false, b1,
                                                       (short)0, acc1, false, false);
    }

    auto store_tile = [&](const v8f& acc, int nbase) {
        int n = nbase + (lane & 15);
        if (n < N) {
            float bval = (epi == 1) ? bias[n] : 0.f;
            #pragma unroll
            for (int r = 0; r < 8; ++r) {
                int m = m0 + ((lane >> 4) << 3) + r;
                float x = acc[r];
                if (epi == 1) {
                    x += bval;
                    x = (x > 20.f) ? x : log1pf(__expf(x));
                }
                C[(size_t)m * N + n] = x;
            }
        }
    };
    store_tile(acc0, n0);
    store_tile(acc1, n0 + 16);
}

// ---------------- weight prep: f32 (N,K) -> bf16 K-major padded --------------
__global__ void prep_weight(const float* __restrict__ W, __bf16* __restrict__ Wt,
                            int N, int K, int Npad, int Kpad)
{
    int idx = blockIdx.x * blockDim.x + threadIdx.x;
    if (idx >= Kpad * Npad) return;
    int k = idx / Npad, n = idx % Npad;
    float v = 0.f;
    if (n < N && k < K) v = W[(size_t)n * K + k];
    Wt[idx] = (__bf16)v;
}

// ------- DFT basis init: Hann window folded in, bf16 K-major [512][NB_PAD] ---
__global__ void init_basis(__bf16* __restrict__ bcT, __bf16* __restrict__ bsT)
{
    int i = blockIdx.x * blockDim.x + threadIdx.x;
    if (i >= PATCH * NB_PAD) return;
    int k = i / NB_PAD, n = i % NB_PAD;
    float c = 0.f, s = 0.f;
    if (n < NFREQ) {
        float win = 0.5f * (1.f - __cosf((float)k * (2.f * PI_F / 512.f)));
        int r = (n * k) & 511;
        float ang = (float)r * (2.f * PI_F / 512.f);
        c =  win * __cosf(ang);
        s = -win * __sinf(ang);
    }
    bcT[i] = (__bf16)c;
    bsT[i] = (__bf16)s;
}

// ---------------- spectrogram: |rfft(hann * patches)| via 2 WMMA GEMMs -------
// Window is pre-folded into the basis, so A staging is pure loads + pack.
__global__ __launch_bounds__(128)
void dft_wmma(const float* __restrict__ src, const __bf16* __restrict__ bcT,
              const __bf16* __restrict__ bsT, float* __restrict__ spec)
{
    __shared__ unsigned int AsU[16][17];
    const int wave = threadIdx.x >> 5;
    const int lane = threadIdx.x & 31;
    const int m0 = blockIdx.x * 16;                 // over MROWS
    const int n0 = blockIdx.y * 64 + wave * 16;     // over NFREQ (padded basis)

    v8f accR = {}, accI = {};
    #pragma unroll 1
    for (int ks = 0; ks < 16; ++ks) {
        const int kb = ks << 5;
        __syncthreads();
        {
            int i = threadIdx.x;
            #pragma unroll
            for (int it = 0; it < 2; ++it, i += 128) {
                int r = i >> 4, cp = i & 15;
                int m = m0 + r;
                int b = m >> 10, t = m & 1023;
                const float* sp = src + (size_t)b * NSAMP + t * HOP + kb + cp * 2;
                AsU[r][cp] = pack2bf(sp[0], sp[1]);
            }
        }
        __syncthreads();

        v16bf af  = load_a_frag(AsU, lane);
        const size_t brow = (size_t)kb * NB_PAD + n0;
        v16bf bfc = load_b_frag(bcT + brow, NB_PAD, lane);
        v16bf bfs = load_b_frag(bsT + brow, NB_PAD, lane);

        accR = __builtin_amdgcn_wmma_f32_16x16x32_bf16(false, af, false, bfc,
                                                       (short)0, accR, false, false);
        accI = __builtin_amdgcn_wmma_f32_16x16x32_bf16(false, af, false, bfs,
                                                       (short)0, accI, false, false);
    }

    const int n = n0 + (lane & 15);
    if (n < NFREQ) {
        #pragma unroll
        for (int r = 0; r < 8; ++r) {
            int m = m0 + ((lane >> 4) << 3) + r;
            float re = accR[r], im = accI[r];
            spec[(size_t)m * NFREQ + n] = sqrtf(re * re + im * im);
        }
    }
}

// ---------------- split bins into src1 (137..256) / src2 (0..136) ------------
__global__ void split_spec(const float* __restrict__ spec,
                           float* __restrict__ s1, float* __restrict__ s2)
{
    int idx = blockIdx.x * blockDim.x + threadIdx.x;
    if (idx >= MROWS * NFREQ) return;
    int f = idx % NFREQ;
    size_t r = idx / NFREQ;
    float v = spec[idx];
    if (f >= 137) s1[r * 120 + (f - 137)] = v;
    else          s2[r * 137 + f] = v;
}

// ---------------- depthwise causal conv (k=4) + bias + SiLU ------------------
__global__ void conv_silu(const float* __restrict__ xz, const float* __restrict__ cw,
                          const float* __restrict__ cb, float* __restrict__ u, int di)
{
    int idx = blockIdx.x * blockDim.x + threadIdx.x;
    if (idx >= MROWS * di) return;
    int c = idx % di;
    int bl = idx / di;
    int t = bl & 1023, b = bl >> 10;
    float s = cb[c];
    #pragma unroll
    for (int j = 0; j < 4; ++j) {
        int tt = t - 3 + j;
        if (tt >= 0) s += cw[c * 4 + j] * xz[(size_t)(b * TSTEPS + tt) * (2 * di) + c];
    }
    u[idx] = s / (1.f + __expf(-s));
}

// ---------------- selective scan: one thread per (b, channel) ----------------
__global__ void scan_kernel(const float* __restrict__ delta, const float* __restrict__ u,
                            const float* __restrict__ xdbl, const float* __restrict__ xz,
                            const float* __restrict__ A_log, const float* __restrict__ Dp,
                            float* __restrict__ y, int di, int dtr)
{
    int idx = blockIdx.x * blockDim.x + threadIdx.x;
    if (idx >= BATCH * di) return;
    int d = idx % di, b = idx / di;
    float Arow[16], h[16];
    #pragma unroll
    for (int n = 0; n < 16; ++n) { Arow[n] = -__expf(A_log[d * 16 + n]); h[n] = 0.f; }
    const float Dd = Dp[d];
    const int xw = dtr + 32;
    for (int t = 0; t < TSTEPS; ++t) {
        size_t row = (size_t)(b * TSTEPS + t);
        float dt = delta[row * di + d];
        float ut = u[row * di + d];
        const float* Bt = xdbl + row * xw + dtr;
        const float* Ct = Bt + 16;
        float dtu = dt * ut;
        float acc = 0.f;
        #pragma unroll
        for (int n = 0; n < 16; ++n) {
            float hn = __expf(dt * Arow[n]) * h[n] + dtu * Bt[n];
            h[n] = hn;
            acc += hn * Ct[n];
        }
        float zt = xz[row * (size_t)(2 * di) + di + d];
        y[row * di + d] = (acc + ut * Dd) * (zt / (1.f + __expf(-zt)));
    }
}

// ---------------- residual + LayerNorm: io += LN(mo) -------------------------
__global__ __launch_bounds__(256)
void ln_residual(float* __restrict__ io, const float* __restrict__ mo,
                 const float* __restrict__ w, const float* __restrict__ bb, int d)
{
    __shared__ float red[256];
    const size_t row = blockIdx.x;
    const float* xr = mo + row * d;
    float* orow = io + row * d;

    float s = 0.f;
    for (int i = threadIdx.x; i < d; i += 256) s += xr[i];
    red[threadIdx.x] = s; __syncthreads();
    for (int o = 128; o > 0; o >>= 1) {
        if (threadIdx.x < o) red[threadIdx.x] += red[threadIdx.x + o];
        __syncthreads();
    }
    float mu = red[0] / (float)d;
    __syncthreads();

    float v = 0.f;
    for (int i = threadIdx.x; i < d; i += 256) { float t = xr[i] - mu; v += t * t; }
    red[threadIdx.x] = v; __syncthreads();
    for (int o = 128; o > 0; o >>= 1) {
        if (threadIdx.x < o) red[threadIdx.x] += red[threadIdx.x + o];
        __syncthreads();
    }
    float rstd = rsqrtf(red[0] / (float)d + 1e-5f);

    for (int i = threadIdx.x; i < d; i += 256)
        orow[i] += (xr[i] - mu) * rstd * w[i] + bb[i];
}

// ---------------- gated concat: [a*src1, (1-a)*src2] -------------------------
__global__ void combine(const float* __restrict__ s1, const float* __restrict__ s2,
                        const float* __restrict__ alpha, float* __restrict__ comb)
{
    int idx = blockIdx.x * blockDim.x + threadIdx.x;
    if (idx >= MROWS * NFREQ) return;
    int j = idx % NFREQ;
    size_t r = idx / NFREQ;
    float a = 1.f / (1.f + __expf(-alpha[0]));
    comb[idx] = (j < 120) ? a * s1[r * 120 + j] : (1.f - a) * s2[r * 137 + (j - 120)];
}

// ---------------- mean over T + classifier -----------------------------------
__global__ __launch_bounds__(512)
void mean_cls(const float* __restrict__ x, const float* __restrict__ cw,
              const float* __restrict__ cb, float* __restrict__ out)
{
    __shared__ float xm[NFREQ];
    int b = blockIdx.x;
    for (int f = threadIdx.x; f < NFREQ; f += 512) {
        float s = 0.f;
        for (int t = 0; t < TSTEPS; ++t)
            s += x[((size_t)b * TSTEPS + t) * NFREQ + f];
        xm[f] = s * (1.f / (float)TSTEPS);
    }
    __syncthreads();
    if (threadIdx.x < 5) {
        float s = cb[threadIdx.x];
        for (int f = 0; f < NFREQ; ++f) s += xm[f] * cw[threadIdx.x * NFREQ + f];
        out[b * 5 + threadIdx.x] = s;
    }
}

// ============================================================
// Host orchestration
// ============================================================
static inline int padN(int n) { return (n + 127) & ~127; }  // N pad: mult of 128
static inline int padK(int k) { return (k + 31) & ~31; }    // K pad: mult of 32

struct MambaP {
    const float *in_w, *conv_w, *conv_b, *xproj_w, *dt_w, *dt_b, *A_log, *D, *out_w;
    int d, di, dtr;
};

static void prep(hipStream_t s, const float* W, __bf16* Wt, int N, int K) {
    int Np = padN(N), Kp = padK(K);
    int tot = Np * Kp;
    prep_weight<<<(tot + 255) / 256, 256, 0, s>>>(W, Wt, N, K, Np, Kp);
}

static void gemm(hipStream_t s, const float* A, const __bf16* Wt, const float* bias,
                 float* C, int M, int N, int K, int lda, int epi) {
    gemm_bf16_wmma<<<dim3(M / 16, (N + 127) / 128), 128, 0, s>>>(
        A, Wt, bias, C, M, N, K, lda, padN(N), epi);
}

static void run_mamba(hipStream_t s, float* x_io, const MambaP& p,
                      const float* ln_w, const float* ln_b,
                      float* xz, float* u, float* xdbl, float* delta,
                      float* ybuf, float* mo,
                      __bf16* wt_in, __bf16* wt_xp, __bf16* wt_dt, __bf16* wt_out)
{
    const int M = MROWS;
    const int twoDi = 2 * p.di, xw = p.dtr + 32;

    prep(s, p.in_w,    wt_in,  twoDi, p.d);
    prep(s, p.xproj_w, wt_xp,  xw,    p.di);
    prep(s, p.dt_w,    wt_dt,  p.di,  p.dtr);
    prep(s, p.out_w,   wt_out, p.d,   p.di);

    gemm(s, x_io, wt_in, nullptr, xz, M, twoDi, p.d, p.d, 0);

    int tot = M * p.di;
    conv_silu<<<(tot + 255) / 256, 256, 0, s>>>(xz, p.conv_w, p.conv_b, u, p.di);

    gemm(s, u,    wt_xp, nullptr, xdbl,  M, xw,   p.di,  p.di, 0);
    gemm(s, xdbl, wt_dt, p.dt_b,  delta, M, p.di, p.dtr, xw,   1);

    int nch = BATCH * p.di;
    scan_kernel<<<(nch + 127) / 128, 128, 0, s>>>(
        delta, u, xdbl, xz, p.A_log, p.D, ybuf, p.di, p.dtr);

    gemm(s, ybuf, wt_out, nullptr, mo, M, p.d, p.di, p.di, 0);

    ln_residual<<<M, 256, 0, s>>>(x_io, mo, ln_w, ln_b, p.d);
}

extern "C" void kernel_launch(void* const* d_in, const int* in_sizes, int n_in,
                              void* d_out, int out_size, void* d_ws, size_t ws_size,
                              hipStream_t stream)
{
    (void)in_sizes; (void)n_in; (void)out_size; (void)ws_size;
    const float* src   = (const float*)d_in[0];
    const float* alpha = (const float*)d_in[1];
    auto mk = [&](int base, int d) {
        MambaP p;
        p.in_w    = (const float*)d_in[base + 0];
        p.conv_w  = (const float*)d_in[base + 1];
        p.conv_b  = (const float*)d_in[base + 2];
        p.xproj_w = (const float*)d_in[base + 3];
        p.dt_w    = (const float*)d_in[base + 4];
        p.dt_b    = (const float*)d_in[base + 5];
        p.A_log   = (const float*)d_in[base + 6];
        p.D       = (const float*)d_in[base + 7];
        p.out_w   = (const float*)d_in[base + 8];
        p.d = d; p.di = 2 * d; p.dtr = (d + 15) / 16;
        return p;
    };
    MambaP m1 = mk(2, 120),  m2 = mk(11, 120);
    MambaP m3 = mk(20, 137), m4 = mk(29, 137);
    MambaP m5 = mk(38, 257), m6 = mk(47, 257);
    const float* ln1_w = (const float*)d_in[56];
    const float* ln1_b = (const float*)d_in[57];
    const float* ln2_w = (const float*)d_in[58];
    const float* ln2_b = (const float*)d_in[59];
    const float* ln_w  = (const float*)d_in[60];
    const float* ln_b  = (const float*)d_in[61];
    const float* cls_w = (const float*)d_in[62];
    const float* cls_b = (const float*)d_in[63];
    float* out = (float*)d_out;

    // ---- workspace carve-up ----
    char* wsp = (char*)d_ws;
    auto carve = [&](size_t bytes) {
        char* p = wsp; wsp += (bytes + 255) & ~(size_t)255; return p;
    };
    const int DI_MAX = 514, XW_MAX = 49;
    __bf16* bcT   = (__bf16*)carve((size_t)PATCH * NB_PAD * 2);
    __bf16* bsT   = (__bf16*)carve((size_t)PATCH * NB_PAD * 2);
    // max padded weight tiles (sized for the d=257 layer)
    __bf16* wt_in  = (__bf16*)carve((size_t)padK(257) * padN(1028) * 2); // 288*1152
    __bf16* wt_xp  = (__bf16*)carve((size_t)padK(514) * padN(49)   * 2); // 544*128
    __bf16* wt_dt  = (__bf16*)carve((size_t)padK(17)  * padN(514)  * 2); // 32*640
    __bf16* wt_out = (__bf16*)carve((size_t)padK(514) * padN(257)  * 2); // 544*384
    float*  spec  = (float*)carve((size_t)MROWS * NFREQ * 4);   // reused as comb
    float*  src1  = (float*)carve((size_t)MROWS * 120 * 4);
    float*  src2  = (float*)carve((size_t)MROWS * 137 * 4);
    float*  xz    = (float*)carve((size_t)MROWS * 2 * DI_MAX * 4);
    float*  u     = (float*)carve((size_t)MROWS * DI_MAX * 4);
    float*  xdbl  = (float*)carve((size_t)MROWS * XW_MAX * 4);
    float*  delta = (float*)carve((size_t)MROWS * DI_MAX * 4);
    float*  ybuf  = (float*)carve((size_t)MROWS * DI_MAX * 4);
    float*  mo    = (float*)carve((size_t)MROWS * NFREQ * 4);
    float*  comb  = spec;

    // 1) DFT basis + spectrogram (|rfft| via two bf16 WMMA GEMMs)
    init_basis<<<(PATCH * NB_PAD + 255) / 256, 256, 0, stream>>>(bcT, bsT);
    dft_wmma<<<dim3(MROWS / 16, (NFREQ + 63) / 64), 128, 0, stream>>>(src, bcT, bsT, spec);

    // 2) band split
    int tot = MROWS * NFREQ;
    split_spec<<<(tot + 255) / 256, 256, 0, stream>>>(spec, src1, src2);

    // 3) branch Mamba stacks
    run_mamba(stream, src1, m1, ln1_w, ln1_b, xz, u, xdbl, delta, ybuf, mo,
              wt_in, wt_xp, wt_dt, wt_out);
    run_mamba(stream, src1, m2, ln1_w, ln1_b, xz, u, xdbl, delta, ybuf, mo,
              wt_in, wt_xp, wt_dt, wt_out);
    run_mamba(stream, src2, m3, ln2_w, ln2_b, xz, u, xdbl, delta, ybuf, mo,
              wt_in, wt_xp, wt_dt, wt_out);
    run_mamba(stream, src2, m4, ln2_w, ln2_b, xz, u, xdbl, delta, ybuf, mo,
              wt_in, wt_xp, wt_dt, wt_out);

    // 4) gated concat + fused Mamba stack
    combine<<<(tot + 255) / 256, 256, 0, stream>>>(src1, src2, alpha, comb);
    run_mamba(stream, comb, m5, ln_w, ln_b, xz, u, xdbl, delta, ybuf, mo,
              wt_in, wt_xp, wt_dt, wt_out);
    run_mamba(stream, comb, m6, ln_w, ln_b, xz, u, xdbl, delta, ybuf, mo,
              wt_in, wt_xp, wt_dt, wt_out);

    // 5) temporal mean + classifier
    mean_cls<<<BATCH, 512, 0, stream>>>(comb, cls_w, cls_b, out);
}
